// BioPlausibleSNN_22101901705930
// MI455X (gfx1250) — compile-verified, matrix operand order checked by reading, not run on hardware
//
#include <hip/hip_runtime.h>
#include <hip/hip_bf16.h>
#include <stdint.h>
#include <stddef.h>

// ---------------------------------------------------------------------------
// BioPlausible SNN forward for gfx1250 (MI455X), wave32 + WMMA bf16.
//
// Pipeline:
//   snn_zero  : zero spike ping-pong buffers + grid-barrier words.
//   snn_split : one-time fully-parallel fp32 -> bf16 hi/lo split of x
//               (moves all input-conversion VALU work out of the serialized
//                scan loop; bandwidth-bound: ~128MB @ 23.3TB/s ≈ 5.5us).
//   snn_scan  : ONE persistent kernel runs all T=256 steps. grid = 32 blocks
//               (one per 16-hidden-column slice), 512 threads = 16 wave32
//               (one 16-row batch tile each). Weights staged to LDS as bf16
//               hi/lo ONCE; all neuron state (mem/syn/pre/post/elig/refr/y)
//               is register-resident for the whole scan (8 cells per lane).
//               Steps separated by one device-wide atomic barrier (32 blocks
//               are trivially co-resident); spike array is double-buffered.
//   snn_out   : out = (y/T) . W_out^T + b_out in one GEMM (time-mean folded
//               through the linear output layer).
//
// Numerics: forward spikes are exactly {0,1} (surrogate == hard in fwd), so
// the dominant recurrent GEMM uses exact-bf16 A with a 2-term bf16 hi/lo
// split of W_rec (~fp32 accuracy). Input/output GEMMs use 3-term splits
// (hi*hi + hi*lo + lo*hi).
//
// WMMA fragment layouts per CDNA5 ISA 7.12.2:
//   A 16x32 bf16: lane m=L%16; v16bf slot i <-> k = 16*(i/8) + (L>=16?8:0)+i%8
//   B 32x16 bf16: assumed mirror with lane n=L%16 (same k formula).
//   C/D f32: lane n=L%16; element e <-> row m = e + (L>=16?8:0).
// ---------------------------------------------------------------------------

typedef __attribute__((ext_vector_type(16))) __bf16 v16bf;
typedef __attribute__((ext_vector_type(8)))  __bf16 v8bf;
typedef __attribute__((ext_vector_type(4)))  __bf16 v4bf;
typedef __attribute__((ext_vector_type(8)))  float  v8f;
typedef __attribute__((ext_vector_type(4)))  float  v4f;

#define SNN_B 256
#define SNN_T 256
#define SNN_D 256
#define SNN_H 512
#define SNN_O 128
#define SCAN_BLOCKS (SNN_H / 16)  // 32

constexpr float K_DM    = 0.95122942450071400f;  // exp(-1/20)
constexpr float K_DS    = 0.88249690258459546f;  // exp(-1/8)
constexpr float K_DPRE  = 0.95122942450071400f;  // exp(-1/20)
constexpr float K_DPOST = 0.95122942450071400f;  // exp(-1/20)
constexpr float K_DELIG = 0.99004983374916811f;  // exp(-1/100)
constexpr float K_VTH   = 1.0f;
constexpr float K_VRST  = 0.0f;
constexpr float K_AP    = 0.01f;
constexpr float K_AM    = 0.012f;
constexpr float K_REFR  = 2.0f;
constexpr float K_HTGT  = 0.008f;  // 8 Hz * dt/1000
constexpr float K_HREG  = 0.002f;

__device__ __forceinline__ v8f wmma_bf16(v16bf a, v16bf b, v8f c) {
  return __builtin_amdgcn_wmma_f32_16x16x32_bf16(
      false, a, false, b, (short)0, c, false, false);
}

__device__ __forceinline__ v16bf cat8(v8bf lo, v8bf hi) {
  return __builtin_shufflevector(lo, hi, 0, 1, 2, 3, 4, 5, 6, 7,
                                 8, 9, 10, 11, 12, 13, 14, 15);
}

__device__ __forceinline__ void split16(const float* f, v16bf& hi, v16bf& lo) {
#pragma unroll
  for (int i = 0; i < 16; ++i) {
    __bf16 h = (__bf16)f[i];
    hi[i] = h;
    lo[i] = (__bf16)(f[i] - (float)h);
  }
}

// Device-wide barrier, monotonically increasing target (no sense reversal).
__device__ __forceinline__ void grid_barrier(int* cnt, int* flag, int target) {
  __threadfence();  // make this block's spike stores device-visible
  __syncthreads();
  if (threadIdx.x == 0) {
    int prev = __hip_atomic_fetch_add(cnt, 1, __ATOMIC_ACQ_REL,
                                      __HIP_MEMORY_SCOPE_AGENT);
    if (prev == SCAN_BLOCKS - 1) {
      __hip_atomic_store(cnt, 0, __ATOMIC_RELAXED, __HIP_MEMORY_SCOPE_AGENT);
      __hip_atomic_store(flag, target, __ATOMIC_RELEASE,
                         __HIP_MEMORY_SCOPE_AGENT);
    } else {
      while (__hip_atomic_load(flag, __ATOMIC_ACQUIRE,
                               __HIP_MEMORY_SCOPE_AGENT) < target) {
        __builtin_amdgcn_s_sleep(2);
      }
    }
  }
  __syncthreads();
}

// ---------------------------------------------------------------------------
__global__ void snn_zero(uint32_t* __restrict__ p, int n) {
  int i = blockIdx.x * blockDim.x + threadIdx.x;
  if (i < n) p[i] = 0u;
}

// One-time fp32 -> (bf16 hi, bf16 lo) split of x. 4 elements per thread.
__global__ void snn_split(const float* __restrict__ x, __bf16* __restrict__ xh,
                          __bf16* __restrict__ xl, int n4) {
  int i = blockIdx.x * blockDim.x + threadIdx.x;
  if (i >= n4) return;
  v4f f = *(const v4f*)(x + (size_t)i * 4);
  v4bf h, l;
#pragma unroll
  for (int c = 0; c < 4; ++c) {
    __bf16 hv = (__bf16)f[c];
    h[c] = hv;
    l[c] = (__bf16)(f[c] - (float)hv);
  }
  *(v4bf*)(xh + (size_t)i * 4) = h;
  *(v4bf*)(xl + (size_t)i * 4) = l;
}

// ---------------------------------------------------------------------------
__global__ __launch_bounds__(512)
void snn_scan(const __bf16* __restrict__ xh,   // [B,T,D] bf16 hi
              const __bf16* __restrict__ xl,   // [B,T,D] bf16 lo
              const float* __restrict__ W_in,  // [H,D]
              const float* __restrict__ b_in,  // [H]
              const float* __restrict__ W_rec, // [H,H]
              const float* __restrict__ b_rec, // [H]
              float* __restrict__ yacc,        // [B,H] out: sum_t(spk+0.1elig)
              __bf16* __restrict__ spk0,       // [B,H] ping
              __bf16* __restrict__ spk1,       // [B,H] pong
              int* __restrict__ bar_cnt, int* __restrict__ bar_flag) {
  constexpr int WI_STR = SNN_D + 8;  // 264-halves column stride (pad)
  constexpr int WR_STR = SNN_H + 8;  // 520
  __shared__ __align__(16) __bf16 sWiH[16 * WI_STR];
  __shared__ __align__(16) __bf16 sWiL[16 * WI_STR];
  __shared__ __align__(16) __bf16 sWrH[16 * WR_STR];
  __shared__ __align__(16) __bf16 sWrL[16 * WR_STR];
  __shared__ float sColSum[16 * 16];
  __shared__ float sAdj[16];

  const int tid    = threadIdx.x;
  const int wid    = tid >> 5;
  const int lane   = tid & 31;
  const int n      = lane & 15;
  const int laneHi = lane >> 4;
  const int h0     = blockIdx.x * 16;
  const int b0     = wid * 16;
  const int mA     = lane & 15;        // A-fragment row this lane loads
  const int bRow   = b0 + laneHi * 8;  // C-fragment rows bRow..bRow+7
  const int hcol   = h0 + n;

  // ---- Stage weight slices to LDS once (bf16 hi/lo, column-major [n][k]).
  for (int i = tid; i < 16 * SNN_D; i += 512) {
    int nn = i >> 8, k = i & (SNN_D - 1);
    float w  = W_in[(size_t)(h0 + nn) * SNN_D + k];
    __bf16 h = (__bf16)w;
    sWiH[nn * WI_STR + k] = h;
    sWiL[nn * WI_STR + k] = (__bf16)(w - (float)h);
  }
  for (int i = tid; i < 16 * SNN_H; i += 512) {
    int nn = i >> 9, k = i & (SNN_H - 1);
    float w  = W_rec[(size_t)(h0 + nn) * SNN_H + k];
    __bf16 h = (__bf16)w;
    sWrH[nn * WR_STR + k] = h;
    sWrL[nn * WR_STR + k] = (__bf16)(w - (float)h);
  }
  __syncthreads();

  // ---- Whole-scan register-resident neuron state (8 cells per lane).
  float r_mem[8] = {}, r_syn[8] = {}, r_pre[8] = {}, r_post[8] = {};
  float r_elig[8] = {}, r_refr[8] = {}, r_y[8] = {};

  const float   bsum  = b_in[hcol] + b_rec[hcol];
  const __bf16* xhb   = xh + (size_t)(b0 + mA) * (SNN_T * SNN_D);
  const __bf16* xlb   = xl + (size_t)(b0 + mA) * (SNN_T * SNN_D);
  const __bf16* row0  = spk0 + (size_t)(b0 + mA) * SNN_H;
  const __bf16* row1  = spk1 + (size_t)(b0 + mA) * SNN_H;

  for (int t = 0; t < SNN_T; ++t) {
    const __bf16* srow  = (t & 1) ? row1 : row0;  // spikes from step t-1
    __bf16*       snext = (t & 1) ? spk0 : spk1;  // spikes of step t

    v8f acc = {0.f, 0.f, 0.f, 0.f, 0.f, 0.f, 0.f, 0.f};

    // Input GEMM: x[b,t,:] . W_in[h,:]  (pre-split bf16, 3-term)
    {
      const __bf16* xhr = xhb + (size_t)t * SNN_D;
      const __bf16* xlr = xlb + (size_t)t * SNN_D;
      for (int kb = 0; kb < SNN_D; kb += 32) {
        const int ao = kb + laneHi * 8;
        v16bf ah = cat8(*(const v8bf*)(xhr + ao), *(const v8bf*)(xhr + ao + 16));
        v16bf al = cat8(*(const v8bf*)(xlr + ao), *(const v8bf*)(xlr + ao + 16));
        const int o0 = n * WI_STR + ao;
        v16bf bh = cat8(*(const v8bf*)(&sWiH[o0]),
                        *(const v8bf*)(&sWiH[o0 + 16]));
        v16bf bl = cat8(*(const v8bf*)(&sWiL[o0]),
                        *(const v8bf*)(&sWiL[o0 + 16]));
        acc = wmma_bf16(ah, bh, acc);
        acc = wmma_bf16(ah, bl, acc);
        acc = wmma_bf16(al, bh, acc);
      }
    }

    // Recurrent GEMM: spk_prev[b,:] . W_rec[h,:]  (spk exact in bf16)
    for (int kb = 0; kb < SNN_H; kb += 32) {
      const int ao = kb + laneHi * 8;
      v16bf av = cat8(*(const v8bf*)(srow + ao), *(const v8bf*)(srow + ao + 16));
      const int o0 = n * WR_STR + ao;
      v16bf bh = cat8(*(const v8bf*)(&sWrH[o0]),
                      *(const v8bf*)(&sWrH[o0 + 16]));
      v16bf bl = cat8(*(const v8bf*)(&sWrL[o0]),
                      *(const v8bf*)(&sWrL[o0 + 16]));
      acc = wmma_bf16(av, bh, acc);
      acc = wmma_bf16(av, bl, acc);
    }

    // LIF/STDP phase 1 (register state).
    float memv[8], spkv[8];
    float csum = 0.f;
#pragma unroll
    for (int e = 0; e < 8; ++e) {
      float sv = K_DS * r_syn[e] + acc[e] + bsum;
      float mv = K_DM * r_mem[e] + sv;
      float sp = (mv >= K_VTH && r_refr[e] <= 0.f) ? 1.f : 0.f;
      float pr = K_DPRE * r_pre[e] + sp;
      float po = K_DPOST * r_post[e] + sp;
      float el = K_DELIG * r_elig[e] + (K_AP * pr * po - K_AM * po);
      r_syn[e] = sv; r_pre[e] = pr; r_post[e] = po; r_elig[e] = el;
      memv[e] = mv; spkv[e] = sp;
      csum += sp;
    }

    // Homeostasis: batch mean per column (intra-block; block owns full batch).
    csum += __shfl_xor(csum, 16, 32);
    if (lane < 16) sColSum[wid * 16 + lane] = csum;
    __syncthreads();
    if (tid < 16) {
      float tot = 0.f;
#pragma unroll
      for (int w = 0; w < 16; ++w) tot += sColSum[w * 16 + tid];
      sAdj[tid] = K_HREG * (tot * (1.f / SNN_B) - K_HTGT);
    }
    __syncthreads();
    const float adj = sAdj[n];

    // Phase 2: reset/refractory/output accumulator + publish spikes.
#pragma unroll
    for (int e = 0; e < 8; ++e) {
      float sp = spkv[e];
      r_mem[e]  = memv[e] * (1.f - sp) + K_VRST * sp - adj;
      r_refr[e] = (sp > 0.f) ? K_REFR : fmaxf(r_refr[e] - 1.f, 0.f);
      r_y[e]   += sp + 0.1f * r_elig[e];
      snext[(size_t)(bRow + e) * SNN_H + hcol] = (__bf16)sp;
    }

    // One device-wide barrier per step (double buffering covers the rest).
    grid_barrier(bar_cnt, bar_flag, t + 1);
  }

  // Publish the time-summed output accumulator for the final GEMM.
#pragma unroll
  for (int e = 0; e < 8; ++e)
    yacc[(size_t)(bRow + e) * SNN_H + hcol] = r_y[e];
}

// ---------------------------------------------------------------------------
// out[b,o] = (y[b,:]/T) . W_out[o,:] + b_out[o]   (3-term bf16 split GEMM)
__global__ __launch_bounds__(32)
void snn_out(const float* __restrict__ yacc, const float* __restrict__ W_out,
             const float* __restrict__ b_out, float* __restrict__ out) {
  const int lane = threadIdx.x & 31;
  const int n = lane & 15, laneHi = lane >> 4, mA = lane & 15;
  const int o0 = blockIdx.x * 16;
  const int b0 = blockIdx.y * 16;

  v8f acc = {0.f, 0.f, 0.f, 0.f, 0.f, 0.f, 0.f, 0.f};
  const float* yrow = yacc + (size_t)(b0 + mA) * SNN_H;
  const float* wrow = W_out + (size_t)(o0 + n) * SNN_H;

  for (int kb = 0; kb < SNN_H; kb += 32) {
    const v4f* pa0 = (const v4f*)(yrow + kb + laneHi * 8);
    const v4f* pa1 = (const v4f*)(yrow + kb + 16 + laneHi * 8);
    v4f a0 = pa0[0], a1 = pa0[1], a2 = pa1[0], a3 = pa1[1];
    float fa[16] = {a0.x, a0.y, a0.z, a0.w, a1.x, a1.y, a1.z, a1.w,
                    a2.x, a2.y, a2.z, a2.w, a3.x, a3.y, a3.z, a3.w};
    v16bf ah, al;
    split16(fa, ah, al);

    const v4f* pb0 = (const v4f*)(wrow + kb + laneHi * 8);
    const v4f* pb1 = (const v4f*)(wrow + kb + 16 + laneHi * 8);
    v4f c0 = pb0[0], c1 = pb0[1], c2 = pb1[0], c3 = pb1[1];
    float fb[16] = {c0.x, c0.y, c0.z, c0.w, c1.x, c1.y, c1.z, c1.w,
                    c2.x, c2.y, c2.z, c2.w, c3.x, c3.y, c3.z, c3.w};
    v16bf bh, bl;
    split16(fb, bh, bl);

    acc = wmma_bf16(ah, bh, acc);
    acc = wmma_bf16(ah, bl, acc);
    acc = wmma_bf16(al, bh, acc);
  }

  const float bo = b_out[o0 + n];
#pragma unroll
  for (int e = 0; e < 8; ++e) {
    out[(size_t)(b0 + laneHi * 8 + e) * SNN_O + o0 + n] =
        acc[e] * (1.f / SNN_T) + bo;
  }
}

// ---------------------------------------------------------------------------
extern "C" void kernel_launch(void* const* d_in, const int* in_sizes, int n_in,
                              void* d_out, int out_size, void* d_ws,
                              size_t ws_size, hipStream_t stream) {
  const float* x     = (const float*)d_in[0];  // [B,T,D]
  const float* W_in  = (const float*)d_in[1];  // [H,D]
  const float* b_in  = (const float*)d_in[2];  // [H]
  const float* W_rec = (const float*)d_in[3];  // [H,H]
  const float* b_rec = (const float*)d_in[4];  // [H]
  const float* W_out = (const float*)d_in[5];  // [O,H]
  const float* b_out = (const float*)d_in[6];  // [O]
  float*       out   = (float*)d_out;          // [B,O]

  // Scratch: yacc | spk0 | spk1 | barrier | x_hi | x_lo  (~68.2 MB total).
  constexpr size_t SZ    = (size_t)SNN_B * SNN_H * sizeof(float);   // 512 KB
  constexpr size_t SPKSZ = (size_t)SNN_B * SNN_H * sizeof(__bf16);  // 256 KB
  constexpr size_t XN    = (size_t)SNN_B * SNN_T * SNN_D;           // 16.7 M
  char*   ws   = (char*)d_ws;
  float*  yacc = (float*)(ws);
  __bf16* spk0 = (__bf16*)(ws + SZ);
  __bf16* spk1 = (__bf16*)(ws + SZ + SPKSZ);
  int*    bar  = (int*)(ws + SZ + 2 * SPKSZ);  // [0]=count, [1]=flag
  __bf16* xh   = (__bf16*)(ws + SZ + 2 * SPKSZ + 256);
  __bf16* xl   = xh + XN;

  // Zero spike buffers + barrier words (deterministic per call).
  const int zn = (int)((2 * SPKSZ + 16) / 4);
  snn_zero<<<(zn + 255) / 256, 256, 0, stream>>>((uint32_t*)(ws + SZ), zn);

  // One-time parallel bf16 hi/lo split of x (off the critical scan path).
  const int n4 = (int)(XN / 4);
  snn_split<<<(n4 + 255) / 256, 256, 0, stream>>>(x, xh, xl, n4);

  // One persistent kernel runs the whole scan.
  snn_scan<<<SCAN_BLOCKS, 512, 0, stream>>>(xh, xl, W_in, b_in, W_rec, b_rec,
                                            yacc, spk0, spk1, bar, bar + 1);

  // Final fused output GEMM (time-mean folded through the linear layer).
  snn_out<<<dim3(SNN_O / 16, SNN_B / 16), 32, 0, stream>>>(yacc, W_out, b_out,
                                                           out);
}
